// MultiHeadAttention_4690104287263
// MI455X (gfx1250) — matrix operand order repaired and checked by reading, not compiled, not run on hardware
//
#include <hip/hip_runtime.h>
#include <hip/hip_bf16.h>

typedef _Float16 f16;
typedef _Float16 v8h  __attribute__((ext_vector_type(8)));
typedef _Float16 v16h __attribute__((ext_vector_type(16)));
typedef float    v8f  __attribute__((ext_vector_type(8)));

#define WMMA_F32_F16(a, b, c) \
  __builtin_amdgcn_wmma_f32_16x16x32_f16(false, (a), false, (b), (short)0, (c), false, false)

static __device__ __forceinline__ v16h cat8(v8h lo, v8h hi) {
  return __builtin_shufflevector(lo, hi, 0, 1, 2, 3, 4, 5, 6, 7, 8, 9, 10, 11, 12, 13, 14, 15);
}

// ---------------------------------------------------------------------------
// Elementwise fp32 -> f16 convert (activations).
// ---------------------------------------------------------------------------
__global__ void conv_f16_kernel(f16* __restrict__ out, const float* __restrict__ in, int n) {
  int idx = blockIdx.x * blockDim.x + threadIdx.x;
  if (idx < n) out[idx] = (f16)in[idx];
}

// ---------------------------------------------------------------------------
// W[K][N] fp32 -> Wt[Np][Kp] f16 (transposed, zero-padded).
// ---------------------------------------------------------------------------
__global__ void transw_kernel(f16* __restrict__ out, const float* __restrict__ W,
                              int K, int N, int Kp, int Np) {
  int idx = blockIdx.x * blockDim.x + threadIdx.x;
  if (idx >= Np * Kp) return;
  int n = idx / Kp, kk = idx % Kp;
  float v = (n < N && kk < K) ? W[(size_t)kk * N + n] : 0.f;
  out[(size_t)n * Kp + kk] = (f16)v;
}

// ---------------------------------------------------------------------------
// Zero the pad columns [from, to) of each row of ao (stride halves).
// ---------------------------------------------------------------------------
__global__ void zero_tail_kernel(f16* __restrict__ ao, int rows, int stride, int from, int to) {
  int w = to - from;
  int idx = blockIdx.x * blockDim.x + threadIdx.x;
  if (idx >= rows * w) return;
  int r = idx / w, c = from + idx % w;
  ao[(size_t)r * stride + c] = (f16)0.f;
}

// ---------------------------------------------------------------------------
// LDS-free WMMA GEMM: C[M,N] = A[M,Kp] @ Wt[Np,Kp]^T + bias[N]
// A, Wt pre-converted f16 (Kp mult of 32, Np mult of 128, zero-padded;
// M must be mult of 128). Fragments loaded directly from global (b128 pairs).
// Block = 128 threads (4 waves); block tile 128x128; wave tile 64x64
// (4x4 WMMA accumulators -> 16 WMMA per 8 fragment loads per K-step).
// ---------------------------------------------------------------------------
__global__ __launch_bounds__(128)
void gemm_f16_tn(float* __restrict__ C, const f16* __restrict__ A,
                 const f16* __restrict__ Wt, const float* __restrict__ bias,
                 int M, int Kp, int N) {
  const int tid  = threadIdx.x;
  const int lane = tid & 31;
  const int wave = tid >> 5;
  const int wm0  = (wave >> 1) * 64;
  const int wn0  = (wave & 1) * 64;
  const int bm   = blockIdx.y * 128;
  const int bn   = blockIdx.x * 128;

  const int l16  = lane & 15;
  const int lhi  = lane >> 4;   // 0 or 1
  const int klo8 = lhi * 8;
  const int kk16 = lhi * 16;

  // Per-lane row base pointers (A rows for A-fragments, Wt rows for B-fragments)
  const f16* arow[4];
  const f16* brow[4];
#pragma unroll
  for (int i = 0; i < 4; ++i) {
    arow[i] = A  + (size_t)(bm + wm0 + i * 16 + l16) * Kp;
    brow[i] = Wt + (size_t)(bn + wn0 + i * 16 + l16) * Kp;
  }

  v8f acc[4][4] = {};

  const int KT = Kp >> 5;
  for (int kt = 0; kt < KT; ++kt) {
    const int k0 = kt << 5;
    // Prefetch ahead into the cache hierarchy
    __builtin_prefetch(arow[0] + k0 + 64, 0, 1);
    __builtin_prefetch(brow[0] + k0 + 64, 0, 1);

    v16h af[4], bf[4];
#pragma unroll
    for (int i = 0; i < 4; ++i) {
      // A fragment: elems 0..15 -> K = k0+kk16 .. +15 (contiguous in A row)
      af[i] = cat8(*(const v8h*)(arow[i] + k0 + kk16),
                   *(const v8h*)(arow[i] + k0 + kk16 + 8));
      // B fragment: elems 0..15 -> K = k0+kk16 .. +15 (contiguous in Wt row)
      bf[i] = cat8(*(const v8h*)(brow[i] + k0 + kk16),
                   *(const v8h*)(brow[i] + k0 + kk16 + 8));
    }
#pragma unroll
    for (int i = 0; i < 4; ++i)
#pragma unroll
      for (int j = 0; j < 4; ++j)
        acc[i][j] = WMMA_F32_F16(af[i], bf[j], acc[i][j]);
  }

  // Epilogue: C layout — lane holds (m = lhi*8 + r, n = l16) in acc[r]
#pragma unroll
  for (int i = 0; i < 4; ++i) {
#pragma unroll
    for (int j = 0; j < 4; ++j) {
      int gn = bn + wn0 + j * 16 + l16;
      if (gn >= N) continue;
      float bb = bias[gn];
#pragma unroll
      for (int r = 0; r < 8; ++r) {
        int gm = bm + wm0 + i * 16 + lhi * 8 + r;
        if (gm < M) C[(size_t)gm * N + gn] = acc[i][j][r] + bb;
      }
    }
  }
}

// ---------------------------------------------------------------------------
// RoPE: in fp32 (B,S,H*D) -> out f16 (B,H,S,D), D=128
// ---------------------------------------------------------------------------
__global__ void rope_kernel(f16* __restrict__ out, const float* __restrict__ in,
                            int Bn, int Sn, int Hn, int D) {
  int idx = blockIdx.x * blockDim.x + threadIdx.x;
  int half = D >> 1;
  int total = Bn * Sn * Hn * half;
  if (idx >= total) return;
  int j = idx % half;
  int h = (idx / half) % Hn;
  int s = (idx / (half * Hn)) % Sn;
  int b = idx / (half * Hn * Sn);
  float invf = __expf(-(float)j / (float)half * 9.210340371976184f);  // 10000^(-j/half)
  float ang = (float)s * invf;
  float cs = __cosf(ang), sn = __sinf(ang);
  const float* base = in + (size_t)(b * Sn + s) * (Hn * D) + h * D;
  float x1 = base[j], x2 = base[half + j];
  f16* ob = out + (size_t)((b * Hn + h) * Sn + s) * D;
  ob[j]        = (f16)(x1 * cs - x2 * sn);
  ob[half + j] = (f16)(x2 * cs + x1 * sn);
}

// ---------------------------------------------------------------------------
// SwiGLU gate: vp fp32 (B,S,2*DV*H) -> vg f16 (B,H,S,DVP) zero-padded 85->96
// ---------------------------------------------------------------------------
__global__ void swiglu_kernel(f16* __restrict__ vg, const float* __restrict__ vp,
                              int Bn, int Sn, int Hn, int DVv, int DVp) {
  int idx = blockIdx.x * blockDim.x + threadIdx.x;
  int total = Bn * Hn * Sn * DVp;
  if (idx >= total) return;
  int j = idx % DVp;
  int s = (idx / DVp) % Sn;
  int h = (idx / (DVp * Sn)) % Hn;
  int b = idx / (DVp * Sn * Hn);
  float r = 0.f;
  if (j < DVv) {
    const float* row = vp + (size_t)(b * Sn + s) * (2 * DVv * Hn);
    float a = row[h * DVv + j];
    float g = row[DVv * Hn + h * DVv + j];
    r = a * (g / (1.f + __expf(-g)));
  }
  vg[(size_t)((b * Hn + h) * Sn + s) * DVp + j] = (f16)r;
}

// ---------------------------------------------------------------------------
// Flash attention: one wave (32 threads) per (b, h, 16-row q tile).
// qr/kr f16 (B,H,S,128), vg f16 (B,H,S,96). ao f16 (B,S,stride=1376).
// ---------------------------------------------------------------------------
__global__ __launch_bounds__(32)
void attn_kernel(f16* __restrict__ ao, const f16* __restrict__ qr,
                 const f16* __restrict__ kr, const f16* __restrict__ vg) {
  constexpr int S = 2048, H = 16, DQK = 128, DVP = 96, DVv = 85, OST = 1376;
  __shared__ __attribute__((aligned(16))) f16 Plds[16 * 40];  // P tile 16x32, stride 40
  __shared__ __attribute__((aligned(16))) f16 Vt[96 * 40];    // V transposed [dv][key], stride 40

  const int lane  = threadIdx.x;
  const int qt    = blockIdx.x;
  const int h     = blockIdx.y;
  const int b     = blockIdx.z;
  const int qbase = qt * 16;
  const int l16   = lane & 15;
  const int lhi   = lane >> 4;
  const int klo8  = lhi * 8;

  const size_t bh = (size_t)(b * H + h);
  const f16* qP = qr + (bh * S + qbase) * DQK;
  const f16* kP = kr + bh * S * DQK;
  const f16* vP = vg + bh * S * DVP;

  // Resident Q fragments (A layout), 4 chunks over K=128
  v16h qf[4];
#pragma unroll
  for (int c = 0; c < 4; ++c) {
    const f16* row = qP + (size_t)l16 * DQK + c * 32;
    qf[c] = cat8(*(const v8h*)(row + lhi * 16), *(const v8h*)(row + lhi * 16 + 8));
  }

  float mrow[8], lrow[8];
  v8f oacc[6] = {};
#pragma unroll
  for (int r = 0; r < 8; ++r) { mrow[r] = -1e30f; lrow[r] = 0.f; }

  const float scale = 0.08838834764831845f;  // 128^-0.5

  for (int jb = 0; jb <= qbase + 15; jb += 32) {
    // ---- scores S[16q x 32k] = Q @ K^T, contraction over DQK=128 ----
    v8f s0 = {}, s1 = {};
#pragma unroll
    for (int c = 0; c < 4; ++c) {
      const f16* kr0 = kP + (size_t)(jb + l16) * DQK + c * 32 + lhi * 16;
      v16h kf0 = cat8(*(const v8h*)kr0, *(const v8h*)(kr0 + 8));
      s0 = WMMA_F32_F16(qf[c], kf0, s0);
      const f16* kr1 = kP + (size_t)(jb + 16 + l16) * DQK + c * 32 + lhi * 16;
      v16h kf1 = cat8(*(const v8h*)kr1, *(const v8h*)(kr1 + 8));
      s1 = WMMA_F32_F16(qf[c], kf1, s1);
    }

    // ---- online softmax; C layout: lane holds (m = lhi*8+r, n = l16) ----
    const int key0 = jb + l16, key1 = key0 + 16;
#pragma unroll
    for (int r = 0; r < 8; ++r) {
      int qrow = qbase + lhi * 8 + r;
      float a0 = (key0 <= qrow) ? s0[r] * scale : -1e30f;
      float a1 = (key1 <= qrow) ? s1[r] * scale : -1e30f;
      float t = fmaxf(a0, a1);
#pragma unroll
      for (int mk = 1; mk <= 8; mk <<= 1) t = fmaxf(t, __shfl_xor(t, mk, 32));
      float newm = fmaxf(mrow[r], t);
      float p0 = __expf(a0 - newm);
      float p1 = __expf(a1 - newm);
      float rs = p0 + p1;
#pragma unroll
      for (int mk = 1; mk <= 8; mk <<= 1) rs += __shfl_xor(rs, mk, 32);
      float corr = __expf(mrow[r] - newm);
      lrow[r] = lrow[r] * corr + rs;
      mrow[r] = newm;
#pragma unroll
      for (int v = 0; v < 6; ++v) oacc[v][r] *= corr;
      Plds[(lhi * 8 + r) * 40 + l16]      = (f16)p0;
      Plds[(lhi * 8 + r) * 40 + 16 + l16] = (f16)p1;
    }

    // ---- stage V block (32 keys x 96 dv) transposed: Vt[dv][key] ----
    {
      const f16* vrow = vP + (size_t)(jb + lane) * DVP;
#pragma unroll
      for (int t = 0; t < 12; ++t) {
        v8h tmp = *(const v8h*)(vrow + t * 8);
#pragma unroll
        for (int e = 0; e < 8; ++e) Vt[(t * 8 + e) * 40 + lane] = tmp[e];
      }
    }
    __syncthreads();

    // ---- O += P @ V : P as one 16x32 A fragment, V as 6 B fragments ----
    const f16* prow = &Plds[l16 * 40];
    v16h pf = cat8(*(const v8h*)(prow + lhi * 16), *(const v8h*)(prow + lhi * 16 + 8));
#pragma unroll
    for (int v = 0; v < 6; ++v) {
      const f16* vtr = &Vt[(v * 16 + l16) * 40 + lhi * 16];
      v16h vf = cat8(*(const v8h*)vtr, *(const v8h*)(vtr + 8));
      oacc[v] = WMMA_F32_F16(pf, vf, oacc[v]);
    }
    __syncthreads();
  }

  // ---- epilogue: normalize and scatter f16 (B,S,stride 1376) ----
#pragma unroll
  for (int r = 0; r < 8; ++r) {
    float inv = 1.f / lrow[r];
    int qrow = qbase + lhi * 8 + r;
    f16* orow = ao + (size_t)(b * S + qrow) * OST + h * DVv;
#pragma unroll
    for (int v = 0; v < 6; ++v) {
      int dv = v * 16 + l16;
      if (dv < DVv) orow[dv] = (f16)(oacc[v][r] * inv);
    }
  }
}

// ---------------------------------------------------------------------------
extern "C" void kernel_launch(void* const* d_in, const int* in_sizes, int n_in,
                              void* d_out, int out_size, void* d_ws, size_t ws_size,
                              hipStream_t stream) {
  const float* q  = (const float*)d_in[0];
  const float* k  = (const float*)d_in[1];
  const float* v  = (const float*)d_in[2];
  // d_in[3] = mask (bool) — causal mask applied analytically, unused
  const float* Wq = (const float*)d_in[4];
  const float* bq = (const float*)d_in[5];
  const float* Wk = (const float*)d_in[6];
  const float* bk = (const float*)d_in[7];
  const float* Wv = (const float*)d_in[8];
  const float* bv = (const float*)d_in[9];
  const float* Wo = (const float*)d_in[10];
  const float* bo = (const float*)d_in[11];
  float* out = (float*)d_out;

  constexpr int Bc = 2, Sc = 2048, Dm = 2048, Hc = 16, DQK = 128, DVv = 85, DVP = 96;
  constexpr int M    = Bc * Sc;        // 4096 (mult of 128)
  constexpr int Nv   = 2 * DVv * Hc;   // 2720
  constexpr int Nvp  = 2816;           // Nv padded to 128
  constexpr int HDV  = DVv * Hc;       // 1360
  constexpr int HDVp = 1376;           // padded to 32

  char* ws = (char*)d_ws;
  size_t off = 0;
  auto alloc = [&](size_t bytes) -> void* {
    void* p = ws + off;
    off = (off + bytes + 255) & ~(size_t)255;
    return p;
  };
  f16*   act16 = (f16*)alloc((size_t)M * Dm * sizeof(f16));     // q/k/v f16, later ao
  f16*   wbuf  = (f16*)alloc((size_t)Nvp * Dm * sizeof(f16));   // transposed weights (cycled)
  float* qp    = (float*)alloc((size_t)M * Dm * sizeof(float));
  float* kp    = (float*)alloc((size_t)M * Dm * sizeof(float));
  float* vp    = (float*)alloc((size_t)M * Nv * sizeof(float));
  f16*   qrr   = (f16*)alloc((size_t)Bc * Hc * Sc * DQK * sizeof(f16));
  f16*   krr   = (f16*)alloc((size_t)Bc * Hc * Sc * DQK * sizeof(f16));
  f16*   vgb   = (f16*)alloc((size_t)Bc * Hc * Sc * DVP * sizeof(f16));

  const int nact = M * Dm;
  dim3 gblk(128);

  // ---- Q projection ----
  conv_f16_kernel<<<(nact + 255) / 256, 256, 0, stream>>>(act16, q, nact);
  transw_kernel<<<(Dm * Dm + 255) / 256, 256, 0, stream>>>(wbuf, Wq, Dm, Dm, Dm, Dm);
  gemm_f16_tn<<<dim3(Dm / 128, M / 128), gblk, 0, stream>>>(qp, act16, wbuf, bq, M, Dm, Dm);
  // ---- K projection ----
  conv_f16_kernel<<<(nact + 255) / 256, 256, 0, stream>>>(act16, k, nact);
  transw_kernel<<<(Dm * Dm + 255) / 256, 256, 0, stream>>>(wbuf, Wk, Dm, Dm, Dm, Dm);
  gemm_f16_tn<<<dim3(Dm / 128, M / 128), gblk, 0, stream>>>(kp, act16, wbuf, bk, M, Dm, Dm);
  // ---- V projection (N padded 2720->2816 via zero rows in wbuf) ----
  conv_f16_kernel<<<(nact + 255) / 256, 256, 0, stream>>>(act16, v, nact);
  transw_kernel<<<(Nvp * Dm + 255) / 256, 256, 0, stream>>>(wbuf, Wv, Dm, Nv, Dm, Nvp);
  gemm_f16_tn<<<dim3(Nvp / 128, M / 128), gblk, 0, stream>>>(vp, act16, wbuf, bv, M, Dm, Nv);

  // ---- RoPE + SwiGLU ----
  int nrope = Bc * Sc * Hc * (DQK / 2);
  rope_kernel<<<(nrope + 255) / 256, 256, 0, stream>>>(qrr, qp, Bc, Sc, Hc, DQK);
  rope_kernel<<<(nrope + 255) / 256, 256, 0, stream>>>(krr, kp, Bc, Sc, Hc, DQK);
  int nsw = Bc * Hc * Sc * DVP;
  swiglu_kernel<<<(nsw + 255) / 256, 256, 0, stream>>>(vgb, vp, Bc, Sc, Hc, DVv, DVP);

  // ---- Attention (writes f16, stride HDVp) ----
  f16* ao = act16;  // act16 dead after V GEMM; M*HDVp*2 <= M*Dm*2
  attn_kernel<<<dim3(Sc / 16, Hc, Bc), 32, 0, stream>>>(ao, qrr, krr, vgb);
  int ntail = M * (HDVp - HDV);
  zero_tail_kernel<<<(ntail + 255) / 256, 256, 0, stream>>>(ao, M, HDVp, HDV, HDVp);

  // ---- Output projection (K padded 1360->1376 via zero cols in wbuf/ao) ----
  transw_kernel<<<(Dm * HDVp + 255) / 256, 256, 0, stream>>>(wbuf, Wo, HDV, Dm, HDVp, Dm);
  gemm_f16_tn<<<dim3(Dm / 128, M / 128), gblk, 0, stream>>>(out, ao, wbuf, bo, M, HDVp, Dm);
}
